// SC_83726092468887
// MI455X (gfx1250) — compile-verified, matrix operand order checked by reading, not run on hardware
//
#include <hip/hip_runtime.h>
#include <hip/hip_bf16.h>

// ---------------------------------------------------------------------------
// DANet dual-attention head for MI455X (gfx1250, wave32, WMMA).
// All GEMM-shaped math runs through v_wmma_f32_16x16x32_bf16 (K=32 == Ci).
// x and all attention operands are staged once in bf16 (position-major) so
// every WMMA operand is a single 32B vector load; PAM is flash-attention
// style (4096x4096 energy never materialized).
// ---------------------------------------------------------------------------

typedef __attribute__((ext_vector_type(16))) __bf16 v16bf;
typedef __attribute__((ext_vector_type(8)))  float  v8f;

static __device__ __forceinline__ v8f wmma_bf16(v16bf a, v16bf b, v8f c) {
  return __builtin_amdgcn_wmma_f32_16x16x32_bf16(false, a, false, b,
                                                 (short)0, c, false, false);
}

#define B_   4
#define HW_  4096
#define CI_  32

// ======================= weight repack kernels ==============================
// Blob layout: B operand in contiguous k order (k = 16*half + j), same order as the A loads:
//   lane n (0-15) holds column n, K = 0..15 ; lanes 16-31 hold K = 16..31.
// Blob index = ((t*NCHUNK + ch)*NT + nt)*512 + lane*16 + j   (bf16)

__global__ void repack_w3x3_dual_k(const float* __restrict__ w0,
                                   const float* __restrict__ w1,
                                   __bf16* __restrict__ out, int CIN) {
  int idx = blockIdx.x * 256 + threadIdx.x;
  int NCHUNK = CIN / 32;
  if (idx >= 9 * NCHUNK * 4 * 512) return;
  int j = idx & 15, lane = (idx >> 4) & 31, nt = (idx >> 9) & 3;
  int rest = idx >> 11;
  int ch = rest % NCHUNK, t = rest / NCHUNK;
  int n = lane & 15, k = ch * 32 + (lane >> 4) * 16 + j;
  int oc = (nt & 1) * 16 + n;
  const float* w = (nt < 2) ? w0 : w1;
  out[idx] = (__bf16)w[((size_t)oc * CIN + k) * 9 + t];
}

__global__ void repack_w3x3_single_k(const float* __restrict__ w,
                                     __bf16* __restrict__ out, int CIN) {
  int idx = blockIdx.x * 256 + threadIdx.x;
  int NCHUNK = CIN / 32;
  if (idx >= 9 * NCHUNK * 2 * 512) return;
  int j = idx & 15, lane = (idx >> 4) & 31, nt = (idx >> 9) & 1;
  int rest = idx >> 10;
  int ch = rest % NCHUNK, t = rest / NCHUNK;
  int n = lane & 15, k = ch * 32 + (lane >> 4) * 16 + j;
  int oc = nt * 16 + n;
  out[idx] = (__bf16)w[((size_t)oc * CIN + k) * 9 + t];
}

__global__ void repack_w1x1_k(const float* __restrict__ w,
                              __bf16* __restrict__ out) {
  int idx = blockIdx.x * 256 + threadIdx.x;
  if (idx >= 1024) return;
  int j = idx & 15, lane = (idx >> 4) & 31, nt = idx >> 9;
  int n = lane & 15, k = (lane >> 4) * 16 + j;
  int oc = nt * 16 + n;
  out[idx] = (__bf16)w[oc * 32 + k];
}

// ======================= x repack: NCHW f32 -> [B][HW][512] bf16 ============
// Coalesced on both sides via an LDS transpose tile. One-time pass; the
// bf16 image (16MB) + f32 x (33MB) both stay resident in the 192MB L2.
__global__ __launch_bounds__(256)
void repack_x_k(const float* __restrict__ x, __bf16* __restrict__ xbf) {
  int blk = blockIdx.x;                    // B * 16(ctile) * 16(ptile)
  int b = blk >> 8, ct = (blk >> 4) & 15, pt = blk & 15;
  __shared__ float tile[32][257];
  const float* xb = x + ((size_t)b * 512 + ct * 32) * HW_ + pt * 256;
  for (int i = threadIdx.x; i < 32 * 256; i += 256)
    tile[i >> 8][i & 255] = xb[(size_t)(i >> 8) * HW_ + (i & 255)];
  __syncthreads();
  __bf16* ob = xbf + ((size_t)b * HW_ + pt * 256) * 512 + ct * 32;
  for (int i = threadIdx.x; i < 32 * 256; i += 256) {
    int n = i >> 5, c = i & 31;
    ob[(size_t)n * 512 + c] = (__bf16)tile[c][n];   // 64B coalesced stores
  }
}

// ======================= implicit-GEMM conv3x3 ==============================
// Input is bf16 position-major: each A-tile operand is ONE 32B vector load.
// Out-of-bounds taps read a zeroed bf16 row (branch-free halo).
// NT==4: dual weight set (conv1a+conv1b fused, one pass over x).
template <int CIN, int NT>
__global__ __launch_bounds__(256)
void conv3x3_wmma_k(const __bf16* __restrict__ xbf,
                    const __bf16* __restrict__ zrow,
                    const __bf16* __restrict__ wblob,
                    float* __restrict__ y0, float* __restrict__ y1) {
  constexpr int NCHUNK = CIN / 32;
  const int lane = threadIdx.x & 31, wv = threadIdx.x >> 5;
  const int b = blockIdx.x >> 5;                     // 32 blocks / batch
  const int p0 = ((blockIdx.x & 31) * 8 + wv) * 16;  // 128 positions / block
  const int m = lane & 15, half = lane >> 4;
  const int p = p0 + m, h = p >> 6, w = p & 63;
  const __bf16* xb = xbf + (size_t)b * HW_ * CIN;
  v8f acc[NT];
#pragma unroll
  for (int nt = 0; nt < NT; ++nt) acc[nt] = (v8f){};

  for (int t = 0; t < 9; ++t) {
    int dy = t / 3 - 1, dx = t % 3 - 1;
    int hh = h + dy, ww = w + dx;
    bool ok = (hh >= 0) & (hh < 64) & (ww >= 0) & (ww < 64);
    int sp = hh * 64 + ww;
    const __bf16* arow = ok ? (xb + (size_t)sp * CIN + half * 16)
                            : (zrow + half * 16);
    for (int ch = 0; ch < NCHUNK; ++ch) {
      v16bf a = *(const v16bf*)(arow + ch * 32);
      const __bf16* wb = wblob + (size_t)((t * NCHUNK + ch) * NT) * 512;
#pragma unroll
      for (int nt = 0; nt < NT; ++nt) {
        v16bf bm = *(const v16bf*)(wb + nt * 512 + lane * 16);
        acc[nt] = wmma_bf16(a, bm, acc[nt]);
      }
    }
  }
#pragma unroll
  for (int nt = 0; nt < NT; ++nt) {
    float* y;
    int oc;
    if constexpr (NT == 4) { y = (nt < 2) ? y0 : y1; oc = (nt & 1) * 16 + m; }
    else                   { y = y0;                 oc = nt * 16 + m; }
#pragma unroll
    for (int r = 0; r < 8; ++r) {
      int pp = p0 + r + 8 * half;
      y[((size_t)b * CI_ + oc) * HW_ + pp] = acc[nt][r];
    }
  }
}

// ======================= BatchNorm (training mode) ==========================
__global__ __launch_bounds__(256)
void bn_stats_k(const float* __restrict__ xin, float* __restrict__ stats) {
  int c = blockIdx.x;
  float s = 0.f, s2 = 0.f;
  for (int i = threadIdx.x; i < B_ * HW_; i += 256) {
    int b = i >> 12, n = i & 4095;
    float v = xin[((size_t)b * CI_ + c) * HW_ + n];
    s += v; s2 += v * v;
  }
  for (int o = 16; o >= 1; o >>= 1) {
    s += __shfl_xor(s, o, 32);
    s2 += __shfl_xor(s2, o, 32);
  }
  __shared__ float ls[8], ls2[8];
  if ((threadIdx.x & 31) == 0) { ls[threadIdx.x >> 5] = s; ls2[threadIdx.x >> 5] = s2; }
  __syncthreads();
  if (threadIdx.x == 0) {
    float S = 0.f, S2 = 0.f;
    for (int i = 0; i < 8; ++i) { S += ls[i]; S2 += ls2[i]; }
    float mean = S / (float)(B_ * HW_);
    float var = S2 / (float)(B_ * HW_) - mean * mean;
    stats[c] = mean;
    stats[CI_ + c] = rsqrtf(var + 1e-3f);
  }
}

__global__ __launch_bounds__(256)
void bn_apply_k(const float* __restrict__ xin, const float* __restrict__ stats,
                const float* __restrict__ g, const float* __restrict__ bb,
                float* __restrict__ y) {
  int idx = blockIdx.x * 256 + threadIdx.x;   // B*32*4096 total
  int c = (idx >> 12) & 31;
  float v = (xin[idx] - stats[c]) * stats[CI_ + c] * g[c] + bb[c];
  y[idx] = fmaxf(v, 0.f);
}

// ======================= QKV (1x1 conv, WMMA) ===============================
// Outputs in bf16: q,k position-major [B][N][32]; v channel-major [B][32][N]
// so every flash-attention operand load is one contiguous 32B vector.
__global__ __launch_bounds__(256)
void qkv_k(const float* __restrict__ feat,
           const __bf16* __restrict__ wqb, const __bf16* __restrict__ wkb,
           const __bf16* __restrict__ wvb,
           const float* __restrict__ bq, const float* __restrict__ bk,
           const float* __restrict__ bv,
           __bf16* __restrict__ qo, __bf16* __restrict__ ko,
           __bf16* __restrict__ vo) {
  int lane = threadIdx.x & 31, wid = threadIdx.x >> 5;
  int b = blockIdx.x >> 5;
  int p0 = ((blockIdx.x & 31) * 8 + wid) * 16;
  int m = lane & 15, half = lane >> 4, p = p0 + m;
  const float* fb = feat + (size_t)b * CI_ * HW_;
  v16bf a;
#pragma unroll
  for (int j = 0; j < 16; ++j)
    a[j] = (__bf16)fb[(size_t)(half * 16 + j) * HW_ + p];

  {  // Q (position-major bf16)
    v16bf b0 = *(const v16bf*)(wqb + lane * 16);
    v16bf b1 = *(const v16bf*)(wqb + 512 + lane * 16);
    v8f a0 = {}, a1 = {};
    a0 = wmma_bf16(a, b0, a0); a1 = wmma_bf16(a, b1, a1);
    __bf16* ob = qo + (size_t)b * HW_ * CI_;
#pragma unroll
    for (int r = 0; r < 8; ++r) {
      int pp = p0 + r + 8 * half;
      ob[(size_t)pp * CI_ + m]      = (__bf16)(a0[r] + bq[m]);
      ob[(size_t)pp * CI_ + 16 + m] = (__bf16)(a1[r] + bq[16 + m]);
    }
  }
  {  // K (position-major bf16)
    v16bf b0 = *(const v16bf*)(wkb + lane * 16);
    v16bf b1 = *(const v16bf*)(wkb + 512 + lane * 16);
    v8f a0 = {}, a1 = {};
    a0 = wmma_bf16(a, b0, a0); a1 = wmma_bf16(a, b1, a1);
    __bf16* ob = ko + (size_t)b * HW_ * CI_;
#pragma unroll
    for (int r = 0; r < 8; ++r) {
      int pp = p0 + r + 8 * half;
      ob[(size_t)pp * CI_ + m]      = (__bf16)(a0[r] + bk[m]);
      ob[(size_t)pp * CI_ + 16 + m] = (__bf16)(a1[r] + bk[16 + m]);
    }
  }
  {  // V (channel-major bf16)
    v16bf b0 = *(const v16bf*)(wvb + lane * 16);
    v16bf b1 = *(const v16bf*)(wvb + 512 + lane * 16);
    v8f a0 = {}, a1 = {};
    a0 = wmma_bf16(a, b0, a0); a1 = wmma_bf16(a, b1, a1);
    __bf16* ob = vo + (size_t)b * CI_ * HW_;
#pragma unroll
    for (int r = 0; r < 8; ++r) {
      int pp = p0 + r + 8 * half;
      ob[(size_t)m * HW_ + pp]        = (__bf16)(a0[r] + bv[m]);
      ob[(size_t)(16 + m) * HW_ + pp] = (__bf16)(a1[r] + bv[16 + m]);
    }
  }
}

// ======================= PAM flash attention ================================
// Each wave: 16 query rows. Per 32-key step: 2 QK WMMAs -> online softmax
// (16-lane butterflies + v_exp) -> D->A layout fixup via per-wave LDS ->
// 2 PV WMMAs. All operands bf16, single vector load each. Output written
// bf16 position-major (feeds the tail conv directly); residual fused.
#define PROW 48   // LDS row stride (bf16): 96B -> 32B-aligned vector reloads
__global__ __launch_bounds__(256)
void pam_attn_k(const __bf16* __restrict__ qb, const __bf16* __restrict__ kb,
                const __bf16* __restrict__ vb, const float* __restrict__ feat1,
                const float* __restrict__ gamma, __bf16* __restrict__ out) {
  __shared__ __bf16 plds[8][16 * PROW];
  int lane = threadIdx.x & 31, wid = threadIdx.x >> 5;
  int b = blockIdx.x >> 5;
  int n0 = ((blockIdx.x & 31) * 8 + wid) * 16;
  int m = lane & 15, half = lane >> 4;
  const __bf16* qB = qb + (size_t)b * HW_ * CI_;
  const __bf16* kB = kb + (size_t)b * HW_ * CI_;
  const __bf16* vB = vb + (size_t)b * CI_ * HW_;

  v16bf aQ = *(const v16bf*)(qB + (size_t)(n0 + m) * CI_ + half * 16);
  float mrow[8], lrow[8];
#pragma unroll
  for (int r = 0; r < 8; ++r) { mrow[r] = -3.0e38f; lrow[r] = 0.f; }
  v8f o0 = {}, o1 = {};
  __bf16* myp = &plds[wid][0];

  for (int m0 = 0; m0 < HW_; m0 += 32) {
    const __bf16* k0 = kB + (size_t)(m0 + m) * CI_ + half * 16;
    __builtin_prefetch(k0 + 32 * CI_, 0, 1);        // next key tile
    v16bf bK0 = *(const v16bf*)k0;
    v16bf bK1 = *(const v16bf*)(k0 + 16 * CI_);
    v8f s0 = {}, s1 = {};
    s0 = wmma_bf16(aQ, bK0, s0);
    s1 = wmma_bf16(aQ, bK1, s1);

    // online softmax per row (row r lives in VGPR r of each 16-lane half)
#pragma unroll
    for (int r = 0; r < 8; ++r) {
      float sm = fmaxf(s0[r], s1[r]);
      sm = fmaxf(sm, __shfl_xor(sm, 1, 32));
      sm = fmaxf(sm, __shfl_xor(sm, 2, 32));
      sm = fmaxf(sm, __shfl_xor(sm, 4, 32));
      sm = fmaxf(sm, __shfl_xor(sm, 8, 32));
      float newm = fmaxf(mrow[r], sm);
      float scale = __expf(mrow[r] - newm);
      mrow[r] = newm;
      float p0v = __expf(s0[r] - newm);
      float p1v = __expf(s1[r] - newm);
      float ps = p0v + p1v;
      ps += __shfl_xor(ps, 1, 32);
      ps += __shfl_xor(ps, 2, 32);
      ps += __shfl_xor(ps, 4, 32);
      ps += __shfl_xor(ps, 8, 32);
      lrow[r] = lrow[r] * scale + ps;
      o0[r] *= scale; o1[r] *= scale;
      s0[r] = p0v; s1[r] = p1v;
    }
    // D-layout -> A-layout through wave-private LDS
#pragma unroll
    for (int r = 0; r < 8; ++r) {
      int row = r + 8 * half;
      myp[row * PROW + m]      = (__bf16)s0[r];
      myp[row * PROW + 16 + m] = (__bf16)s1[r];
    }
    asm volatile("s_wait_dscnt 0" ::: "memory");    // wave-internal LDS RAW
    v16bf aP = *(const v16bf*)(myp + m * PROW + half * 16);

    const __bf16* v0p = vB + (size_t)m * HW_ + m0 + half * 16;
    v16bf bV0 = *(const v16bf*)v0p;
    v16bf bV1 = *(const v16bf*)(v0p + 16 * HW_);
    o0 = wmma_bf16(aP, bV0, o0);
    o1 = wmma_bf16(aP, bV1, o1);
  }

  float g = gamma[0];
  const float* f1 = feat1 + (size_t)b * CI_ * HW_;
  __bf16* ob = out + (size_t)b * HW_ * CI_;
#pragma unroll
  for (int r = 0; r < 8; ++r) {
    float inv = 1.0f / lrow[r];
    int pp = n0 + r + 8 * half;
    int c0 = m, c1 = 16 + m;
    ob[(size_t)pp * CI_ + c0] = (__bf16)(g * o0[r] * inv + f1[(size_t)c0 * HW_ + pp]);
    ob[(size_t)pp * CI_ + c1] = (__bf16)(g * o1[r] * inv + f1[(size_t)c1 * HW_ + pp]);
  }
}

// ======================= CAM channel attention (32x32, tiny) ================
__global__ void zero_misc_k(float* __restrict__ e, __bf16* __restrict__ zrow) {
  int i = blockIdx.x * 256 + threadIdx.x;
  if (i < B_ * 1024) e[i] = 0.f;
  if (i < 512) zrow[i] = (__bf16)0.f;
}

__global__ __launch_bounds__(256)
void cam_energy_k(const float* __restrict__ feat2, float* __restrict__ energy) {
  int b = blockIdx.x >> 4, chunk = blockIdx.x & 15;   // 256 positions / block
  const float* xf = feat2 + (size_t)b * CI_ * HW_ + chunk * 256;
  __shared__ float tile[32][257];
  for (int i = threadIdx.x; i < 32 * 256; i += 256)
    tile[i >> 8][i & 255] = xf[(size_t)(i >> 8) * HW_ + (i & 255)];
  __syncthreads();
  float* eb = energy + b * 1024;
  for (int pair = threadIdx.x; pair < 1024; pair += 256) {
    int c = pair >> 5, d = pair & 31;
    float s = 0.f;
    for (int n = 0; n < 256; ++n) s += tile[c][n] * tile[d][n];
    atomicAdd(&eb[pair], s);
  }
}

__global__ void cam_softmax_k(const float* __restrict__ energy,
                              float* __restrict__ att) {
  int b = blockIdx.x, c = threadIdx.x;   // 32 threads
  const float* e = energy + b * 1024 + c * 32;
  float mn = 3e38f;
  for (int d = 0; d < 32; ++d) mn = fminf(mn, e[d]);
  // softmax(rowmax - e) == exp(rowmin - e) / sum
  float buf[32], s = 0.f;
  for (int d = 0; d < 32; ++d) { buf[d] = __expf(mn - e[d]); s += buf[d]; }
  float inv = 1.0f / s;
  for (int d = 0; d < 32; ++d) att[b * 1024 + c * 32 + d] = buf[d] * inv;
}

// Output written bf16 position-major (feeds tail conv directly).
__global__ __launch_bounds__(256)
void cam_out_k(const float* __restrict__ att, const float* __restrict__ feat2,
               const float* __restrict__ gamma, __bf16* __restrict__ out) {
  int b = blockIdx.x >> 4, n = (blockIdx.x & 15) * 256 + threadIdx.x;
  __shared__ float a[32][32];
  for (int i = threadIdx.x; i < 1024; i += 256) a[i >> 5][i & 31] = att[b * 1024 + i];
  __syncthreads();
  const float* xf = feat2 + (size_t)b * CI_ * HW_;
  float xv[32];
#pragma unroll
  for (int d = 0; d < 32; ++d) xv[d] = xf[(size_t)d * HW_ + n];
  float g = gamma[0];
  __bf16* ob = out + (size_t)b * HW_ * CI_ + (size_t)n * CI_;
#pragma unroll
  for (int c = 0; c < 32; ++c) {
    float s = 0.f;
#pragma unroll
    for (int d = 0; d < 32; ++d) s += a[c][d] * xv[d];
    ob[c] = (__bf16)(g * s + xv[c]);    // 64B contiguous per thread
  }
}

// ======================= final: sum + 1x1 conv + bias + ReLU ================
__global__ __launch_bounds__(256)
void final_k(const float* __restrict__ sa, const float* __restrict__ sc,
             const __bf16* __restrict__ w2b, const float* __restrict__ b2,
             float* __restrict__ out) {
  int lane = threadIdx.x & 31, wid = threadIdx.x >> 5;
  int b = blockIdx.x >> 5;
  int p0 = ((blockIdx.x & 31) * 8 + wid) * 16;
  int m = lane & 15, half = lane >> 4, p = p0 + m;
  const float* sab = sa + (size_t)b * CI_ * HW_;
  const float* scb = sc + (size_t)b * CI_ * HW_;
  v16bf a;
#pragma unroll
  for (int j = 0; j < 16; ++j) {
    size_t off = (size_t)(half * 16 + j) * HW_ + p;
    a[j] = (__bf16)(sab[off] + scb[off]);
  }
  v16bf b0 = *(const v16bf*)(w2b + lane * 16);
  v16bf b1 = *(const v16bf*)(w2b + 512 + lane * 16);
  v8f a0 = {}, a1 = {};
  a0 = wmma_bf16(a, b0, a0);
  a1 = wmma_bf16(a, b1, a1);
  float* ob = out + (size_t)b * CI_ * HW_;
#pragma unroll
  for (int r = 0; r < 8; ++r) {
    int pp = p0 + r + 8 * half;
    ob[(size_t)m * HW_ + pp]        = fmaxf(a0[r] + b2[m], 0.f);
    ob[(size_t)(16 + m) * HW_ + pp] = fmaxf(a1[r] + b2[16 + m], 0.f);
  }
}

// ======================= host-side launch ===================================
extern "C" void kernel_launch(void* const* d_in, const int* in_sizes, int n_in,
                              void* d_out, int out_size, void* d_ws, size_t ws_size,
                              hipStream_t stream) {
  const float* x    = (const float*)d_in[0];
  const float* w1a  = (const float*)d_in[1];
  const float* g1a  = (const float*)d_in[2];
  const float* b1a  = (const float*)d_in[3];
  const float* w1b  = (const float*)d_in[4];
  const float* g1b  = (const float*)d_in[5];
  const float* b1b  = (const float*)d_in[6];
  const float* wq   = (const float*)d_in[7];
  const float* bq   = (const float*)d_in[8];
  const float* wk   = (const float*)d_in[9];
  const float* bk   = (const float*)d_in[10];
  const float* wvw  = (const float*)d_in[11];
  const float* bv   = (const float*)d_in[12];
  const float* gsa  = (const float*)d_in[13];
  const float* gca  = (const float*)d_in[14];
  const float* w1c  = (const float*)d_in[15];
  const float* g1c  = (const float*)d_in[16];
  const float* b1c  = (const float*)d_in[17];
  const float* w1d  = (const float*)d_in[18];
  const float* g1d  = (const float*)d_in[19];
  const float* b1d  = (const float*)d_in[20];
  const float* w2   = (const float*)d_in[21];
  const float* b2   = (const float*)d_in[22];

  // ---- workspace layout ----
  float* ws = (float*)d_ws;
  const size_t S = (size_t)B_ * CI_ * HW_;     // 524288
  float* f0 = ws + 0 * S;       // conv1a raw
  float* f1 = ws + 1 * S;       // conv1b raw
  float* f2 = ws + 2 * S;       // feat1 -> sa_out
  float* f3 = ws + 3 * S;       // feat2 -> sc_out
  float* f4 = ws + 4 * S;       // conv1c raw
  float* f5 = ws + 5 * S;       // conv1d raw
  float* stats  = ws + 6 * S;   // 4 stages x 64
  float* energy = stats + 256;  // 4x1024
  float* att    = energy + 4096;
  __bf16* xbf  = (__bf16*)(att + 4096);        // [B][HW][512] bf16, 32B-aligned
  __bf16* qbf  = xbf + (size_t)B_ * HW_ * 512; // [B][N][32]
  __bf16* kbf  = qbf + S;                      // [B][N][32]
  __bf16* vbf  = kbf + S;                      // [B][32][N]
  __bf16* sabf = vbf + S;                      // sa_feat [B][N][32]
  __bf16* scbf = sabf + S;                     // sc_feat [B][N][32]
  __bf16* zrow = scbf + S;                     // 512 zeros (conv halo)
  __bf16* wfused = zrow + 512;
  __bf16* wcb = wfused + 9 * 16 * 4 * 512;
  __bf16* wdb = wcb + 9 * 2 * 512;
  __bf16* wqb = wdb + 9 * 2 * 512;
  __bf16* wkb = wqb + 1024;
  __bf16* wvb = wkb + 1024;
  __bf16* w2b = wvb + 1024;

  // ---- one-time staging: weights + x -> bf16 WMMA layouts ----
  repack_w3x3_dual_k<<<(9 * 16 * 4 * 512 + 255) / 256, 256, 0, stream>>>(w1a, w1b, wfused, 512);
  repack_w3x3_single_k<<<(9 * 2 * 512 + 255) / 256, 256, 0, stream>>>(w1c, wcb, 32);
  repack_w3x3_single_k<<<(9 * 2 * 512 + 255) / 256, 256, 0, stream>>>(w1d, wdb, 32);
  repack_w1x1_k<<<4, 256, 0, stream>>>(wq, wqb);
  repack_w1x1_k<<<4, 256, 0, stream>>>(wk, wkb);
  repack_w1x1_k<<<4, 256, 0, stream>>>(wvw, wvb);
  repack_w1x1_k<<<4, 256, 0, stream>>>(w2, w2b);
  zero_misc_k<<<16, 256, 0, stream>>>(energy, zrow);
  repack_x_k<<<B_ * 256, 256, 0, stream>>>(x, xbf);

  // ---- conv1a + conv1b fused (single pass over x) ----
  conv3x3_wmma_k<512, 4><<<B_ * 32, 256, 0, stream>>>(xbf, zrow, wfused, f0, f1);
  bn_stats_k<<<32, 256, 0, stream>>>(f0, stats + 0);
  bn_stats_k<<<32, 256, 0, stream>>>(f1, stats + 64);
  bn_apply_k<<<(int)(S / 256), 256, 0, stream>>>(f0, stats + 0,  g1a, b1a, f2);
  bn_apply_k<<<(int)(S / 256), 256, 0, stream>>>(f1, stats + 64, g1b, b1b, f3);

  // ---- PAM branch ----
  qkv_k<<<B_ * 32, 256, 0, stream>>>(f2, wqb, wkb, wvb, bq, bk, bv, qbf, kbf, vbf);
  pam_attn_k<<<B_ * 32, 256, 0, stream>>>(qbf, kbf, vbf, f2, gsa, sabf);

  // ---- CAM branch ----
  cam_energy_k<<<B_ * 16, 256, 0, stream>>>(f3, energy);
  cam_softmax_k<<<B_, 32, 0, stream>>>(energy, att);
  cam_out_k<<<B_ * 16, 256, 0, stream>>>(att, f3, gca, scbf);

  // ---- tail convs + BN + fusion ----
  conv3x3_wmma_k<32, 2><<<B_ * 32, 256, 0, stream>>>(sabf, zrow, wcb, f4, nullptr);
  conv3x3_wmma_k<32, 2><<<B_ * 32, 256, 0, stream>>>(scbf, zrow, wdb, f5, nullptr);
  bn_stats_k<<<32, 256, 0, stream>>>(f4, stats + 128);
  bn_stats_k<<<32, 256, 0, stream>>>(f5, stats + 192);
  bn_apply_k<<<(int)(S / 256), 256, 0, stream>>>(f4, stats + 128, g1c, b1c, f2);
  bn_apply_k<<<(int)(S / 256), 256, 0, stream>>>(f5, stats + 192, g1d, b1d, f3);
  final_k<<<B_ * 32, 256, 0, stream>>>(f2, f3, w2b, b2, (float*)d_out);
}
